// LlamaBlock_2911987826749
// MI455X (gfx1250) — matrix operand order repaired and checked by reading, not compiled, not run on hardware
//
#include <hip/hip_runtime.h>
#include <cstdint>
#include <cstddef>

#define EMBED    2048
#define HEADS    32
#define HEAD_DIM 64
#define MLP_DIM  8192
#define NEGV     -1e30f

typedef __attribute__((ext_vector_type(16))) __bf16        bf16x16;
typedef __attribute__((ext_vector_type(8)))  float         f32x8;
typedef __attribute__((ext_vector_type(4)))  int           i32x4;
typedef __attribute__((ext_vector_type(8)))  int           i32x8;
typedef __attribute__((ext_vector_type(4)))  unsigned int  u32x4;

union Frag16 { bf16x16 v; i32x4 i[2]; unsigned short u[16]; };
union Acc8   { f32x8 v; float f[8]; };

__device__ __forceinline__ unsigned short f2bf(float f) {
  unsigned int u = __builtin_bit_cast(unsigned int, f);
  u += 0x7fffu + ((u >> 16) & 1u);
  return (unsigned short)(u >> 16);
}
__device__ __forceinline__ float bf2f(unsigned short h) {
  unsigned int u = ((unsigned int)h) << 16;
  return __builtin_bit_cast(float, u);
}

#if __has_builtin(__builtin_amdgcn_tensor_load_to_lds) && \
    __has_builtin(__builtin_amdgcn_s_wait_tensorcnt)
#define USE_TDM 1
#else
#define USE_TDM 0
#endif

// Definitive probe, DEVICE pass only: the host pass never declares amdgcn
// builtins, so asserting there is meaningless (and was the round-3 failure).
#if defined(__HIP_DEVICE_COMPILE__)
static_assert(USE_TDM == 1, "TDM builtins unavailable in device compilation");
#endif

#if USE_TDM
// ---------------------------------------------------------------------------
// TDM descriptor for a 64-row x 64-col bf16 tile, row stride K elements,
// with LDS padding of 4 DWORDs (16B) after every 32 DWORDs (128B = one row)
// so rows land at a 144-byte stride (== ushort[72] row) for bank-spread,
// 16B-aligned ds fragment reads. D# layout per CDNA5 ISA.
// ---------------------------------------------------------------------------
__device__ __forceinline__ void tdm_load_tile64x64(unsigned ldsOff,
                                                   const unsigned short* g,
                                                   unsigned Kstride) {
  unsigned long long ga = (unsigned long long)(size_t)g;
  u32x4 g0 = (u32x4){
      1u,                                   // count=1, user mode, no gather
      ldsOff,                               // lds_addr (bytes)
      (unsigned)ga,                         // global_addr[31:0]
      (unsigned)((ga >> 32) & 0x1ffffffu) | (2u << 30)  // addr[56:32] | type=2
  };
  i32x8 g1;
  g1[0] = (int)((1u << 16) |                // data_size = 1 (2 bytes)
                (1u << 20) |                // pad_enable
                (4u << 22) |                // pad_interval: 32 DWORDs (128B)
                (3u << 25));                // pad_amount:   4 DWORDs (16B)
  g1[1] = (int)((Kstride & 0xffffu) << 16);            // tensor_dim0[15:0]
  g1[2] = (int)((Kstride >> 16) | (64u << 16));        // dim0[31:16]|dim1[15:0]
  g1[3] = (int)(64u << 16);                            // dim1 hi=0 | tile_dim0=64
  g1[4] = (int)64u;                                    // tile_dim1=64, tile_dim2=0
  g1[5] = (int)Kstride;                                // tensor_dim0_stride lo
  g1[6] = 0;
  g1[7] = 0;                                           // dim1_stride unused (2D)
  i32x4 z4 = (i32x4){0, 0, 0, 0};
#if defined(__clang_major__) && __clang_major__ >= 23
  i32x8 z8 = (i32x8){0, 0, 0, 0, 0, 0, 0, 0};
  __builtin_amdgcn_tensor_load_to_lds(g0, g1, z4, z4, z8, 0);
#else
  __builtin_amdgcn_tensor_load_to_lds(g0, g1, z4, z4, 0);
#endif
}
#endif  // USE_TDM

__device__ __forceinline__ unsigned ldsOffset(const void* p) {
  return (unsigned)(size_t)p;  // flat LDS aperture: low 32 bits = LDS offset
}

// ---------------------------------------------------------------------------
// fp32 (K x N) -> bf16 transposed (N x K), LDS-tiled 32x32 for coalescing.
// Transposed weights let TDM stage both GEMM operands as identical row-major
// 64x64 tiles (TDM has no transpose path).
// ---------------------------------------------------------------------------
__global__ __launch_bounds__(256)
void transpose_cast_bf16(const float* __restrict__ in,
                         unsigned short* __restrict__ out, int K, int N) {
  __shared__ float t[32][33];
  const int kb = blockIdx.y * 32, nb = blockIdx.x * 32;
  const int tx = threadIdx.x, ty = threadIdx.y;   // block (32, 8)
  #pragma unroll
  for (int a = ty; a < 32; a += 8)
    t[a][tx] = in[(size_t)(kb + a) * N + nb + tx];
  __syncthreads();
  #pragma unroll
  for (int a = ty; a < 32; a += 8)
    out[(size_t)(nb + a) * K + kb + tx] = f2bf(t[tx][a]);
}

// ---------------------------------------------------------------------------
// RMSNorm (one block per token row) -> bf16
// ---------------------------------------------------------------------------
__global__ __launch_bounds__(256)
void rmsnorm_bf16(const float* __restrict__ x, const float* __restrict__ g,
                  unsigned short* __restrict__ out, int E) {
  __shared__ float red[8];
  const int row = blockIdx.x;
  const float* xr = x + (size_t)row * E;
  float s = 0.f;
  for (int i = threadIdx.x; i < E; i += 256) { float v = xr[i]; s += v * v; }
  #pragma unroll
  for (int off = 16; off > 0; off >>= 1) s += __shfl_xor(s, off, 32);
  if ((threadIdx.x & 31) == 0) red[threadIdx.x >> 5] = s;
  __syncthreads();
  if (threadIdx.x < 8) {
    float t = red[threadIdx.x];
    #pragma unroll
    for (int off = 4; off > 0; off >>= 1) t += __shfl_xor(t, off, 8);
    if (threadIdx.x == 0) red[0] = t;
  }
  __syncthreads();
  const float inv = rsqrtf(red[0] / (float)E + 1e-5f);
  for (int i = threadIdx.x; i < E; i += 256)
    out[(size_t)row * E + i] = f2bf(xr[i] * inv * g[i]);
}

// ---------------------------------------------------------------------------
// bf16 GEMM: C[M,N](f32) = A[M,K] * Bt[N,K]^T (+ residual).
// 128 threads = 4 waves; block tile 64x64; wave tile 32x32 (2x2 WMMA accs).
// TDM double-buffered staging of K=64 slabs: wave 0 DMAs the next slab while
// all waves run 8 WMMAs on the current one.
// ---------------------------------------------------------------------------
template <bool RES>
__global__ __launch_bounds__(128)
void gemm_bf16_wmma(const unsigned short* __restrict__ A,
                    const unsigned short* __restrict__ Bt,
                    const float* __restrict__ residual,
                    float* __restrict__ C, int M, int N, int K) {
  __shared__ __attribute__((aligned(16))) unsigned short As[2][64][72];
  __shared__ __attribute__((aligned(16))) unsigned short Bs[2][64][72];

  const int tid  = threadIdx.x;
  const int wave = tid >> 5;
  const int lane = tid & 31;
  const int lrow = lane & 15;
  const int hi   = lane >> 4;
  const int blockM = blockIdx.y * 64;
  const int blockN = blockIdx.x * 64;
  const int wm = (wave >> 1) * 32;
  const int wn = (wave &  1) * 32;

  const unsigned short* Abase = A  + (size_t)blockM * K;
  const unsigned short* Bbase = Bt + (size_t)blockN * K;

  Acc8 acc[2][2];
  #pragma unroll
  for (int mi = 0; mi < 2; ++mi)
    #pragma unroll
    for (int ni = 0; ni < 2; ++ni)
      acc[mi][ni].v = (f32x8){0.f,0.f,0.f,0.f,0.f,0.f,0.f,0.f};

  auto stage = [&](int bufI, int k0) {
#if USE_TDM
    if (wave == 0) {
      tdm_load_tile64x64(ldsOffset(&As[bufI][0][0]), Abase + k0, (unsigned)K);
      tdm_load_tile64x64(ldsOffset(&Bs[bufI][0][0]), Bbase + k0, (unsigned)K);
    }
#else
    #pragma unroll
    for (int p = 0; p < 4; ++p) {
      int slot = tid + p * 128;            // 512 slots = 64 rows * 8 segs
      int r = slot >> 3, sgi = slot & 7;
      *(i32x4*)&As[bufI][r][sgi * 8] =
          *(const i32x4*)(Abase + (size_t)r * K + k0 + sgi * 8);
      *(i32x4*)&Bs[bufI][r][sgi * 8] =
          *(const i32x4*)(Bbase + (size_t)r * K + k0 + sgi * 8);
    }
#endif
  };

  int buf = 0;
  stage(0, 0);
#if USE_TDM
  if (wave == 0) __builtin_amdgcn_s_wait_tensorcnt(0);
#endif
  __syncthreads();

  for (int k0 = 0; k0 < K; k0 += 64) {
    if (k0 + 64 < K) stage(buf ^ 1, k0 + 64);   // async prefetch of next slab

    #pragma unroll
    for (int ks = 0; ks < 2; ++ks) {
      const int kb = ks * 32;
      Frag16 af[2], bfr[2];
      #pragma unroll
      for (int mi = 0; mi < 2; ++mi) {     // A-frag: row=lane%16, K 0-7 / 16-23
        int r = wm + mi * 16 + lrow;
        af[mi].i[0] = *(const i32x4*)&As[buf][r][kb + hi * 8];
        af[mi].i[1] = *(const i32x4*)&As[buf][r][kb + 16 + hi * 8];
      }
      #pragma unroll
      for (int ni = 0; ni < 2; ++ni) {     // B-frag: col=lane%16, K 0-7 / 16-23
        int n = wn + ni * 16 + lrow;
        bfr[ni].i[0] = *(const i32x4*)&Bs[buf][n][kb + hi * 8];
        bfr[ni].i[1] = *(const i32x4*)&Bs[buf][n][kb + 16 + hi * 8];
      }
      #pragma unroll
      for (int mi = 0; mi < 2; ++mi)
        #pragma unroll
        for (int ni = 0; ni < 2; ++ni)
          acc[mi][ni].v = __builtin_amdgcn_wmma_f32_16x16x32_bf16(
              false, af[mi].v, false, bfr[ni].v, (short)0, acc[mi][ni].v,
              false, false);
    }

#if USE_TDM
    if (wave == 0) __builtin_amdgcn_s_wait_tensorcnt(0);
#endif
    __syncthreads();
    buf ^= 1;
  }

  #pragma unroll
  for (int mi = 0; mi < 2; ++mi)
    #pragma unroll
    for (int ni = 0; ni < 2; ++ni) {
      size_t base = (size_t)(blockM + wm + mi * 16 + hi * 8) * N +
                    (blockN + wn + ni * 16 + lrow);
      float* cp = C + base;
      const float* rp = residual + base;
      #pragma unroll
      for (int j = 0; j < 8; ++j) {
        float v = acc[mi][ni].f[j];
        if (RES) v += rp[(size_t)j * N];
        cp[(size_t)j * N] = v;
      }
    }
}

// ---------------------------------------------------------------------------
// RoPE on q,k + cast & repack q,k,v to [B,H,S,D] bf16
// ---------------------------------------------------------------------------
__global__ void rope_pack(const float* __restrict__ q, const float* __restrict__ k,
                          const float* __restrict__ v,
                          unsigned short* __restrict__ qb, unsigned short* __restrict__ kb,
                          unsigned short* __restrict__ vb, int B, int S) {
  size_t idx = (size_t)blockIdx.x * blockDim.x + threadIdx.x;
  size_t total = (size_t)B * S * HEADS * (HEAD_DIM / 2);
  if (idx >= total) return;
  int d2 = (int)(idx & 31); size_t r = idx >> 5;
  int h = (int)(r % HEADS); r /= HEADS;
  int s = (int)(r % S);     int b = (int)(r / S);
  float inv = powf(500000.0f, -(float)(2 * d2) / (float)HEAD_DIM);
  float ang = (float)s * inv;
  float cs = cosf(ang), sn = sinf(ang);
  size_t src = ((size_t)(b * S + s)) * (HEADS * HEAD_DIM) + (size_t)h * HEAD_DIM;
  size_t dst = (((size_t)b * HEADS + h) * S + s) * HEAD_DIM;
  float q1 = q[src + 2 * d2], q2 = q[src + 2 * d2 + 1];
  float k1 = k[src + 2 * d2], k2 = k[src + 2 * d2 + 1];
  qb[dst + 2 * d2]     = f2bf(q1 * cs - q2 * sn);
  qb[dst + 2 * d2 + 1] = f2bf(q1 * sn + q2 * cs);
  kb[dst + 2 * d2]     = f2bf(k1 * cs - k2 * sn);
  kb[dst + 2 * d2 + 1] = f2bf(k1 * sn + k2 * cs);
  vb[dst + 2 * d2]     = f2bf(v[src + 2 * d2]);
  vb[dst + 2 * d2 + 1] = f2bf(v[src + 2 * d2 + 1]);
}

// ---------------------------------------------------------------------------
// Sparse tile top-k attention. One wave per (b, h, 16-row q-tile).
// ---------------------------------------------------------------------------
__global__ __launch_bounds__(32)
void sparse_attn(const unsigned short* __restrict__ qb,
                 const unsigned short* __restrict__ kb,
                 const unsigned short* __restrict__ vb,
                 unsigned short* __restrict__ out, int B, int S) {
  const int T = S / 16;
  int bid = blockIdx.x;
  const int qt = bid % T; bid /= T;
  const int h  = bid % HEADS;
  const int b  = bid / HEADS;
  const int lane = threadIdx.x;
  const int lrow = lane & 15, hi = lane >> 4;

  __shared__ __attribute__((aligned(16))) unsigned short sc[16][1032];
  __shared__ float tmax[16][64];
  __shared__ unsigned int maskLo[16], maskHi[16];
  __shared__ float rowMax[16], rowSum[16];
  __shared__ __attribute__((aligned(16))) unsigned short vt[64][40];
  __shared__ int tlist[64];
  __shared__ int tcount;

  const size_t headBase = ((size_t)b * HEADS + h) * S;
  const unsigned short* qbase = qb + (headBase + (size_t)qt * 16) * HEAD_DIM;

  Frag16 qf[2];
  #pragma unroll
  for (int f = 0; f < 2; ++f) {
    qf[f].i[0] = *(const i32x4*)(qbase + (size_t)lrow * HEAD_DIM + f * 32 + hi * 8);
    qf[f].i[1] = *(const i32x4*)(qbase + (size_t)lrow * HEAD_DIM + f * 32 + 16 + hi * 8);
  }
  const float scale = 0.125f;

  // ---- pass 1: scores for all causal tiles + per-row tile maxima ----
  for (int kt = 0; kt <= qt; ++kt) {
    const unsigned short* kbase = kb + (headBase + (size_t)kt * 16) * HEAD_DIM;
    Frag16 kf[2];
    #pragma unroll
    for (int f = 0; f < 2; ++f) {
      kf[f].i[0] = *(const i32x4*)(kbase + (size_t)lrow * HEAD_DIM + f * 32 + hi * 8);
      kf[f].i[1] = *(const i32x4*)(kbase + (size_t)lrow * HEAD_DIM + f * 32 + 16 + hi * 8);
    }
    Acc8 a; a.v = (f32x8){0.f,0.f,0.f,0.f,0.f,0.f,0.f,0.f};
    a.v = __builtin_amdgcn_wmma_f32_16x16x32_bf16(false, qf[0].v, false, kf[0].v, (short)0, a.v, false, false);
    a.v = __builtin_amdgcn_wmma_f32_16x16x32_bf16(false, qf[1].v, false, kf[1].v, (short)0, a.v, false, false);
    #pragma unroll
    for (int j = 0; j < 8; ++j) {
      int row = hi * 8 + j;
      int col = lrow;
      float s = a.f[j] * scale;
      if (kt == qt && col > row) s = NEGV;
      sc[row][kt * 16 + col] = f2bf(s);
      float m = s;
      #pragma unroll
      for (int off = 1; off < 16; off <<= 1) m = fmaxf(m, __shfl_xor(m, off, 32));
      if (lrow == 0) tmax[row][kt] = m;
    }
  }
  __syncthreads();

  // ---- pass 2: per-row top-12 tile selection + softmax stats ----
  if (lane < 16) {
    const int row = lane;
    const int nt = qt + 1;
    unsigned long long sel = 0ull;
    for (int pick = 0; pick < 12; ++pick) {
      float best = -3e38f; int bi = -1;
      for (int t = 0; t < nt; ++t) {
        if ((sel >> t) & 1ull) continue;
        float tv = tmax[row][t];
        if (tv > best) { best = tv; bi = t; }
      }
      if (bi >= 0) sel |= 1ull << bi;
    }
    sel |= 1ull << qt;
    float rm = -3e38f;
    for (int t = 0; t < nt; ++t) if ((sel >> t) & 1ull) rm = fmaxf(rm, tmax[row][t]);
    float rs = 0.f;
    for (int t = 0; t < nt; ++t) if ((sel >> t) & 1ull)
      for (int c = 0; c < 16; ++c) rs += __expf(bf2f(sc[row][t * 16 + c]) - rm);
    rowMax[row] = rm; rowSum[row] = rs;
    maskLo[row] = (unsigned int)(sel & 0xffffffffull);
    maskHi[row] = (unsigned int)(sel >> 32);
  }
  __syncthreads();

  if (lane == 0) {
    unsigned long long uni = 0ull;
    for (int r = 0; r < 16; ++r)
      uni |= ((unsigned long long)maskHi[r] << 32) | maskLo[r];
    int ct = 0;
    for (int t = 0; t <= qt; ++t) if ((uni >> t) & 1ull) tlist[ct++] = t;
    tcount = ct;
  }
  __syncthreads();
  const int ct = tcount;
  const unsigned long long myMask =
      ((unsigned long long)maskHi[lrow] << 32) | maskLo[lrow];
  const float myRowMax = rowMax[lrow];

  // ---- pass 3: PV accumulate over union tiles, two tiles per WMMA ----
  Acc8 oacc[4];
  #pragma unroll
  for (int ni = 0; ni < 4; ++ni) oacc[ni].v = (f32x8){0.f,0.f,0.f,0.f,0.f,0.f,0.f,0.f};

  for (int p = 0; p < ct; p += 2) {
    const int ta = tlist[p];
    const int tb = (p + 1 < ct) ? tlist[p + 1] : -1;

    Frag16 pf;
    #pragma unroll
    for (int e = 0; e < 16; ++e) {
      int vg = e >> 1, pp = e & 1;
      int kk = hi * 16 + 2 * vg + pp;
      int t  = (kk < 16) ? ta : tb;
      int c  = kk & 15;
      float val = 0.f;
      if (t >= 0 && ((myMask >> t) & 1ull))
        val = __expf(bf2f(sc[lrow][t * 16 + c]) - myRowMax);
      pf.u[e] = f2bf(val);
    }

    {
      int t = (lane < 16) ? ta : tb;
      int key = (t >= 0) ? t * 16 + lrow : 0;
      const unsigned short* vrow = vb + (headBase + key) * HEAD_DIM;
      #pragma unroll
      for (int sgi = 0; sgi < 8; ++sgi) {
        unsigned short tmp[8];
        *(i32x4*)tmp = *(const i32x4*)(vrow + sgi * 8);
        #pragma unroll
        for (int j = 0; j < 8; ++j)
          vt[sgi * 8 + j][lane] = (t >= 0) ? tmp[j] : (unsigned short)0;
      }
    }
    __syncthreads();

    #pragma unroll
    for (int ni = 0; ni < 4; ++ni) {
      Frag16 vf;
      int d = ni * 16 + lrow;
      vf.i[0] = *(const i32x4*)&vt[d][hi * 16];
      vf.i[1] = *(const i32x4*)&vt[d][hi * 16 + 8];
      oacc[ni].v = __builtin_amdgcn_wmma_f32_16x16x32_bf16(
          false, pf.v, false, vf.v, (short)0, oacc[ni].v, false, false);
    }
    __syncthreads();
  }

  #pragma unroll
  for (int ni = 0; ni < 4; ++ni)
    #pragma unroll
    for (int j = 0; j < 8; ++j) {
      int row = hi * 8 + j;
      int srow = qt * 16 + row;
      float o = oacc[ni].f[j] / rowSum[row];
      out[((size_t)(b * S + srow)) * (HEADS * HEAD_DIM) + (size_t)h * HEAD_DIM +
          ni * 16 + lrow] = f2bf(o);
    }
}

// ---------------------------------------------------------------------------
// SwiGLU elementwise: m = silu(gate) * up  -> bf16
// ---------------------------------------------------------------------------
__global__ void silu_mul(const float* __restrict__ g, const float* __restrict__ u,
                         unsigned short* __restrict__ out, size_t n) {
  size_t i = (size_t)blockIdx.x * blockDim.x + threadIdx.x;
  size_t stride = (size_t)gridDim.x * blockDim.x;
  for (; i < n; i += stride) {
    float gv = g[i];
    float sv = gv / (1.f + __expf(-gv));
    out[i] = f2bf(sv * u[i]);
  }
}

// ---------------------------------------------------------------------------
// Host-side orchestration
// ---------------------------------------------------------------------------
extern "C" void kernel_launch(void* const* d_in, const int* in_sizes, int n_in,
                              void* d_out, int out_size, void* d_ws, size_t ws_size,
                              hipStream_t stream) {
  const float* x      = (const float*)d_in[0];
  const float* anorm  = (const float*)d_in[1];
  const float* wq     = (const float*)d_in[2];
  const float* wk     = (const float*)d_in[3];
  const float* wv     = (const float*)d_in[4];
  const float* wo     = (const float*)d_in[5];
  const float* fnorm  = (const float*)d_in[6];
  const float* wgate  = (const float*)d_in[7];
  const float* wup    = (const float*)d_in[8];
  const float* wdown  = (const float*)d_in[9];

  const int tokens = in_sizes[0] / EMBED;   // B*S
  const int S = 1024;
  const int B = tokens / S;
  const int HD = HEADS * HEAD_DIM;          // 2048

  char* ws = (char*)d_ws;
  size_t off = 0;
  auto alloc = [&](size_t bytes) -> void* {
    void* p = ws + off;
    off += (bytes + 255) & ~(size_t)255;
    return p;
  };
  // all weights stored TRANSPOSED (out_dim x in_dim) in bf16
  unsigned short* wq_t = (unsigned short*)alloc((size_t)EMBED * HD * 2);
  unsigned short* wk_t = (unsigned short*)alloc((size_t)EMBED * HD * 2);
  unsigned short* wv_t = (unsigned short*)alloc((size_t)EMBED * HD * 2);
  unsigned short* wo_t = (unsigned short*)alloc((size_t)HD * EMBED * 2);
  unsigned short* wg_t = (unsigned short*)alloc((size_t)EMBED * MLP_DIM * 2);
  unsigned short* wu_t = (unsigned short*)alloc((size_t)EMBED * MLP_DIM * 2);
  unsigned short* wd_t = (unsigned short*)alloc((size_t)MLP_DIM * EMBED * 2);
  unsigned short* n_b  = (unsigned short*)alloc((size_t)tokens * EMBED * 2);
  float* q_f = (float*)alloc((size_t)tokens * HD * 4);
  float* k_f = (float*)alloc((size_t)tokens * HD * 4);
  float* v_f = (float*)alloc((size_t)tokens * HD * 4);
  unsigned short* qb2 = (unsigned short*)alloc((size_t)tokens * HD * 2);
  unsigned short* kb2 = (unsigned short*)alloc((size_t)tokens * HD * 2);
  unsigned short* vb2 = (unsigned short*)alloc((size_t)tokens * HD * 2);
  unsigned short* at_b = (unsigned short*)alloc((size_t)tokens * HD * 2);
  float* h_f = (float*)alloc((size_t)tokens * EMBED * 4);
  unsigned short* n2_b = (unsigned short*)alloc((size_t)tokens * EMBED * 2);
  float* g_f = (float*)alloc((size_t)tokens * MLP_DIM * 4);
  float* u_f = (float*)alloc((size_t)tokens * MLP_DIM * 4);
  unsigned short* m_b = (unsigned short*)alloc((size_t)tokens * MLP_DIM * 2);

  // 1) weight transpose-casts (one-time per call)
  dim3 tb(32, 8);
  transpose_cast_bf16<<<dim3(HD / 32, EMBED / 32), tb, 0, stream>>>(wq, wq_t, EMBED, HD);
  transpose_cast_bf16<<<dim3(HD / 32, EMBED / 32), tb, 0, stream>>>(wk, wk_t, EMBED, HD);
  transpose_cast_bf16<<<dim3(HD / 32, EMBED / 32), tb, 0, stream>>>(wv, wv_t, EMBED, HD);
  transpose_cast_bf16<<<dim3(EMBED / 32, HD / 32), tb, 0, stream>>>(wo, wo_t, HD, EMBED);
  transpose_cast_bf16<<<dim3(MLP_DIM / 32, EMBED / 32), tb, 0, stream>>>(wgate, wg_t, EMBED, MLP_DIM);
  transpose_cast_bf16<<<dim3(MLP_DIM / 32, EMBED / 32), tb, 0, stream>>>(wup,   wu_t, EMBED, MLP_DIM);
  transpose_cast_bf16<<<dim3(EMBED / 32, MLP_DIM / 32), tb, 0, stream>>>(wdown, wd_t, MLP_DIM, EMBED);

  // 2) attn RMSNorm
  rmsnorm_bf16<<<tokens, 256, 0, stream>>>(x, anorm, n_b, EMBED);

  // 3) QKV projections
  dim3 gqkv(HD / 64, tokens / 64);
  gemm_bf16_wmma<false><<<gqkv, 128, 0, stream>>>(n_b, wq_t, nullptr, q_f, tokens, HD, EMBED);
  gemm_bf16_wmma<false><<<gqkv, 128, 0, stream>>>(n_b, wk_t, nullptr, k_f, tokens, HD, EMBED);
  gemm_bf16_wmma<false><<<gqkv, 128, 0, stream>>>(n_b, wv_t, nullptr, v_f, tokens, HD, EMBED);

  // 4) RoPE + repack to [B,H,S,D] bf16
  {
    size_t total = (size_t)tokens * HEADS * (HEAD_DIM / 2);
    rope_pack<<<(unsigned)((total + 255) / 256), 256, 0, stream>>>(
        q_f, k_f, v_f, qb2, kb2, vb2, B, S);
  }

  // 5) sparse attention
  sparse_attn<<<B * HEADS * (S / 16), 32, 0, stream>>>(qb2, kb2, vb2, at_b, B, S);

  // 6) WO projection + residual
  dim3 gwo(EMBED / 64, tokens / 64);
  gemm_bf16_wmma<true><<<gwo, 128, 0, stream>>>(at_b, wo_t, x, h_f, tokens, EMBED, HD);

  // 7) ffn RMSNorm
  rmsnorm_bf16<<<tokens, 256, 0, stream>>>(h_f, fnorm, n2_b, EMBED);

  // 8) gate/up GEMMs
  dim3 gmlp(MLP_DIM / 64, tokens / 64);
  gemm_bf16_wmma<false><<<gmlp, 128, 0, stream>>>(n2_b, wg_t, nullptr, g_f, tokens, MLP_DIM, EMBED);
  gemm_bf16_wmma<false><<<gmlp, 128, 0, stream>>>(n2_b, wu_t, nullptr, u_f, tokens, MLP_DIM, EMBED);

  // 9) SwiGLU
  silu_mul<<<8192, 256, 0, stream>>>(g_f, u_f, m_b, (size_t)tokens * MLP_DIM);

  // 10) down projection + residual -> d_out (fp32)
  dim3 gdown(EMBED / 64, tokens / 64);
  gemm_bf16_wmma<true><<<gdown, 128, 0, stream>>>(m_b, wd_t, h_f, (float*)d_out,
                                                  tokens, EMBED, MLP_DIM);
}
